// MultiHeadSelfAttention_33174327394384
// MI455X (gfx1250) — matrix operand order repaired and checked by reading, not compiled, not run on hardware
//
#include <hip/hip_runtime.h>
#include <math.h>

typedef __attribute__((ext_vector_type(16))) _Float16 v16h;
typedef __attribute__((ext_vector_type(8)))  _Float16 v8h;
typedef __attribute__((ext_vector_type(8)))  float    v8f;
typedef int v4i __attribute__((vector_size(16)));  // matches builtin signature

#define B_   16
#define T_   512
#define C_   1024
#define H_   16
#define DH_  64
#define F_   3072
#define BT_  8192

#ifndef __has_builtin
#define __has_builtin(x) 0
#endif
#if __has_builtin(__builtin_amdgcn_global_load_async_to_lds_b128)
#define ASYNC_CP 1
#else
#define ASYNC_CP 0
#endif

union HF16 { v16h v; v8h h[2]; };

__device__ __forceinline__ v8f zero8() {
  v8f z;
#pragma unroll
  for (int i = 0; i < 8; ++i) z[i] = 0.0f;
  return z;
}

// A fragment (16x32 f16): lane m = L&15 is the row, half = L>>4.
// elements 0..15 <- K = half*16 + 0..15
__device__ __forceinline__ v16h a_frag(const _Float16* rowp, int half) {
  HF16 u;
  u.h[0] = *reinterpret_cast<const v8h*>(rowp + half * 16);
  u.h[1] = *reinterpret_cast<const v8h*>(rowp + half * 16 + 8);
  return u.v;
}

// B fragment (32x16 f16): lane n = L&15 is the column; p -> 16 contiguous K
// values starting at K = (L>>4)*16 (caller bakes +half*16 into p).
__device__ __forceinline__ v16h b_frag(const _Float16* p) {
  HF16 u;
  u.h[0] = *reinterpret_cast<const v8h*>(p);
  u.h[1] = *reinterpret_cast<const v8h*>(p + 8);
  return u.v;
}

__device__ __forceinline__ v8f wmma_f16(v16h a, v16h b, v8f c) {
  return __builtin_amdgcn_wmma_f32_16x16x32_f16(false, a, false, b, (short)0, c,
                                                false, false);
}

// 16-byte global -> LDS copy. Async (gfx1250 ASYNCcnt path) when available.
// Flat-LDS pointer low 32 bits are the workgroup-relative LDS byte address;
// global flat address == AS1 address, so integer casts are value-correct.
__device__ __forceinline__ void cp16(const _Float16* g, _Float16* l) {
#if ASYNC_CP
  __builtin_amdgcn_global_load_async_to_lds_b128(
      (__attribute__((address_space(1))) v4i*)(unsigned long long)g,
      (__attribute__((address_space(3))) v4i*)(unsigned)(unsigned long long)l,
      0, 0);
#else
  *reinterpret_cast<v8h*>(l) = *reinterpret_cast<const v8h*>(g);
#endif
}

__device__ __forceinline__ void wait_stage() {
#if ASYNC_CP
  asm volatile("s_wait_asynccnt 0" ::: "memory");
#endif
  __syncthreads();
}

#define LDP 72  // padded LDS row stride in halves (multiple of 8 -> 16B aligned)

// stage 128 rows x 64 halves (A) : 1024 b128 chunks / 256 threads = 4 each
__device__ __forceinline__ void stage_a(const _Float16* g, int ldg,
                                        _Float16* l, int t) {
#pragma unroll
  for (int j = 0; j < 4; ++j) {
    const int c = t + j * 256;
    const int row = c >> 3, col = (c & 7) * 8;
    cp16(g + (size_t)row * ldg + col, l + row * LDP + col);
  }
}
// stage 64 rows x 64 halves (B) : 512 chunks / 256 threads = 2 each
__device__ __forceinline__ void stage_b(const _Float16* g, int ldg,
                                        _Float16* l, int t) {
#pragma unroll
  for (int j = 0; j < 2; ++j) {
    const int c = t + j * 256;
    const int row = c >> 3, col = (c & 7) * 8;
    cp16(g + (size_t)row * ldg + col, l + row * LDP + col);
  }
}

// ---------------------------------------------------------------- converts
__global__ void f32_to_f16_kernel(const float* __restrict__ src,
                                  _Float16* __restrict__ dst, int n) {
  int i = blockIdx.x * blockDim.x + threadIdx.x;
  if (i < n) dst[i] = (_Float16)src[i];
}

// ------------------------------------------------- V transpose: VT[b,h,d,t]
__global__ void v_transpose_kernel(const _Float16* __restrict__ qkv,
                                   _Float16* __restrict__ vt) {
  int id = blockIdx.x * blockDim.x + threadIdx.x;  // B*H*DH*T
  int t  = id & (T_ - 1);
  int r  = id >> 9;
  int d  = r & (DH_ - 1);
  int r2 = r >> 6;
  int h  = r2 & (H_ - 1);
  int b  = r2 >> 4;
  vt[id] = qkv[(size_t)(b * T_ + t) * F_ + 2 * C_ + h * DH_ + d];
}

// ---------------- GEMM: Y[M,N] = A[M,K] * W[N,K]^T (+bias), LDS double-buffer
// block = 256 thr (8 waves), tile BM=128 x BN=64, BK=64.
// wave (wm in 0..3, wn in 0..1) owns a 32x32 register tile (2x2 WMMA accs).
template <bool OUT_F16>
__global__ void __launch_bounds__(256)
gemm_lds(const _Float16* __restrict__ A, const _Float16* __restrict__ W,
         void* __restrict__ out, const float* __restrict__ bias, int K, int N) {
  __shared__ _Float16 sA[2][128 * LDP];
  __shared__ _Float16 sB[2][64 * LDP];

  const int t    = threadIdx.x;
  const int lane = t & 31;
  const int m    = lane & 15;
  const int half = lane >> 4;
  const int wave = t >> 5;
  const int wm   = wave & 3;
  const int wn   = wave >> 2;

  const int nblk = N >> 6;
  const int mt0  = (blockIdx.x / nblk) * 128;
  const int nt0  = (blockIdx.x % nblk) * 64;

  const _Float16* ga = A + (size_t)mt0 * K;
  const _Float16* gb = W + (size_t)nt0 * K;

  v8f a00 = zero8(), a01 = zero8(), a10 = zero8(), a11 = zero8();

  stage_a(ga, K, sA[0], t);
  stage_b(gb, K, sB[0], t);

  int buf = 0;
  for (int kb = 0; kb < K; kb += 64) {
    wait_stage();
    if (kb + 64 < K) {
      stage_a(ga + kb + 64, K, sA[buf ^ 1], t);
      stage_b(gb + kb + 64, K, sB[buf ^ 1], t);
    }
    const _Float16* la = sA[buf];
    const _Float16* lb = sB[buf];
#pragma unroll
    for (int ks = 0; ks < 64; ks += 32) {
      v16h af0 = a_frag(la + (wm * 32 + m) * LDP + ks, half);
      v16h af1 = a_frag(la + (wm * 32 + 16 + m) * LDP + ks, half);
      v16h bf0 = b_frag(lb + (wn * 32 + m) * LDP + ks + half * 16);
      v16h bf1 = b_frag(lb + (wn * 32 + 16 + m) * LDP + ks + half * 16);
      a00 = wmma_f16(af0, bf0, a00);
      a01 = wmma_f16(af0, bf1, a01);
      a10 = wmma_f16(af1, bf0, a10);
      a11 = wmma_f16(af1, bf1, a11);
    }
    buf ^= 1;
  }

  const int rb = mt0 + wm * 32 + 8 * half;
  const int cb = nt0 + wn * 32 + m;
#pragma unroll
  for (int i = 0; i < 8; ++i) {
    if constexpr (OUT_F16) {
      _Float16* o0 = (_Float16*)out + (size_t)(rb + i) * N + cb;
      _Float16* o1 = (_Float16*)out + (size_t)(rb + 16 + i) * N + cb;
      o0[0]  = (_Float16)a00[i];
      o0[16] = (_Float16)a01[i];
      o1[0]  = (_Float16)a10[i];
      o1[16] = (_Float16)a11[i];
    } else {
      float* o0 = (float*)out + (size_t)(rb + i) * N + cb;
      float* o1 = (float*)out + (size_t)(rb + 16 + i) * N + cb;
      o0[0]  = a00[i] + bias[cb];
      o0[16] = a01[i] + bias[cb + 16];
      o1[0]  = a10[i] + bias[cb];
      o1[16] = a11[i] + bias[cb + 16];
    }
  }
}

// --------------------------------------------- flash attention, wave32 WMMA
__global__ void __launch_bounds__(128)
attn_kernel(const _Float16* __restrict__ qkv, const _Float16* __restrict__ vt,
            const unsigned char* __restrict__ kpm, _Float16* __restrict__ o16) {
  __shared__ _Float16 plds[4][16 * 32];  // per-wave P tile (16 q x 32 keys)

  const int lane = threadIdx.x & 31;
  const int m    = lane & 15;
  const int half = lane >> 4;
  const int wave = threadIdx.x >> 5;
  const int wid  = blockIdx.x * 4 + wave;
  const int qt   = wid & 31;
  const int h    = (wid >> 5) & 15;
  const int b    = wid >> 9;
  _Float16* pl   = plds[wave];

  const _Float16* qrow = qkv + (size_t)(b * T_ + qt * 16 + m) * F_ + h * DH_;
  const v16h qf0 = a_frag(qrow, half);
  const v16h qf1 = a_frag(qrow + 32, half);

  const _Float16* kb_ = qkv + (size_t)b * T_ * F_ + C_ + h * DH_;  // + key*F_
  const _Float16* vtb = vt + (size_t)((b * H_ + h) * DH_) * T_;    // + d*T_+key
  const unsigned char* pmb = kpm + b * T_;

  v8f o0 = zero8(), o1 = zero8(), o2 = zero8(), o3 = zero8();
  float mrow[8], lrow[8];
#pragma unroll
  for (int i = 0; i < 8; ++i) { mrow[i] = -1e30f; lrow[i] = 0.0f; }

  const int qbase = qt * 16 + 8 * half;
  const int npair = (qt >> 1) + 1;

  for (int p = 0; p < npair; ++p) {
    const int key0 = p * 32 + m;
    const int key1 = key0 + 16;
    const _Float16* kr0 = kb_ + (size_t)key0 * F_ + half * 16;
    const _Float16* kr1 = kb_ + (size_t)key1 * F_ + half * 16;
    __builtin_prefetch(kr0 + (size_t)32 * F_, 0, 0);  // next key pair

    v8f s0 = zero8(), s1 = zero8();
    s0 = wmma_f16(qf0, b_frag(kr0),      s0);
    s0 = wmma_f16(qf1, b_frag(kr0 + 32), s0);
    s1 = wmma_f16(qf0, b_frag(kr1),      s1);
    s1 = wmma_f16(qf1, b_frag(kr1 + 32), s1);

    const bool pm0 = pmb[key0] != 0;
    const bool pm1 = pmb[key1] != 0;

#pragma unroll
    for (int i = 0; i < 8; ++i) {
      const int q = qbase + i;
      float v0 = s0[i] * 0.125f;  // 1/sqrt(Dh)
      float v1 = s1[i] * 0.125f;
      if (key0 > q || pm0) v0 = -1e30f;
      if (key1 > q || pm1) v1 = -1e30f;

      float tv = fmaxf(v0, v1);
      tv = fmaxf(tv, __shfl_xor(tv, 1));
      tv = fmaxf(tv, __shfl_xor(tv, 2));
      tv = fmaxf(tv, __shfl_xor(tv, 4));
      tv = fmaxf(tv, __shfl_xor(tv, 8));
      const float mn = fmaxf(mrow[i], tv);
      const float sc = __expf(mrow[i] - mn);
      const float p0 = __expf(v0 - mn);
      const float p1 = __expf(v1 - mn);
      float rs = p0 + p1;
      rs += __shfl_xor(rs, 1);
      rs += __shfl_xor(rs, 2);
      rs += __shfl_xor(rs, 4);
      rs += __shfl_xor(rs, 8);
      lrow[i] = lrow[i] * sc + rs;
      mrow[i] = mn;
      o0[i] *= sc; o1[i] *= sc; o2[i] *= sc; o3[i] *= sc;

      pl[(i + 8 * half) * 32 + m]      = (_Float16)p0;
      pl[(i + 8 * half) * 32 + 16 + m] = (_Float16)p1;
    }
    asm volatile("s_wait_dscnt 0" ::: "memory");

    HF16 pf;
    pf.h[0] = *reinterpret_cast<const v8h*>(pl + m * 32 + half * 16);
    pf.h[1] = *reinterpret_cast<const v8h*>(pl + m * 32 + half * 16 + 8);

    const _Float16* vp = vtb + (size_t)m * T_ + p * 32 + half * 16;
    o0 = wmma_f16(pf.v, b_frag(vp),                   o0);
    o1 = wmma_f16(pf.v, b_frag(vp + (size_t)16 * T_), o1);
    o2 = wmma_f16(pf.v, b_frag(vp + (size_t)32 * T_), o2);
    o3 = wmma_f16(pf.v, b_frag(vp + (size_t)48 * T_), o3);
  }

  _Float16* orow = o16 + (size_t)(b * T_ + qbase) * C_ + h * DH_ + m;
#pragma unroll
  for (int i = 0; i < 8; ++i) {
    const float inv = 1.0f / lrow[i];
    _Float16* op = orow + (size_t)i * C_;
    op[0]  = (_Float16)(o0[i] * inv);
    op[16] = (_Float16)(o1[i] * inv);
    op[32] = (_Float16)(o2[i] * inv);
    op[48] = (_Float16)(o3[i] * inv);
  }
}

extern "C" void kernel_launch(void* const* d_in, const int* in_sizes, int n_in,
                              void* d_out, int out_size, void* d_ws, size_t ws_size,
                              hipStream_t stream) {
  (void)in_sizes; (void)n_in; (void)out_size; (void)ws_size;
  const float* x           = (const float*)d_in[0];
  /* d_in[1] = tril mask: causal structure applied analytically */
  const unsigned char* kpm = (const unsigned char*)d_in[2];
  const float* w_qkv       = (const float*)d_in[3];
  const float* w_out       = (const float*)d_in[4];
  const float* b_out       = (const float*)d_in[5];
  float* out               = (float*)d_out;

  char* ws = (char*)d_ws;
  _Float16* X16   = (_Float16*)(ws);                       // 16 MB
  _Float16* WQ16  = (_Float16*)(ws + ((size_t)16 << 20));  //  6 MB
  _Float16* WO16  = (_Float16*)(ws + ((size_t)23 << 20));  //  2 MB
  _Float16* QKV16 = (_Float16*)(ws + ((size_t)25 << 20));  // 48 MB
  _Float16* VT16  = (_Float16*)(ws + ((size_t)73 << 20));  // 16 MB
  _Float16* O16   = (_Float16*)(ws + ((size_t)89 << 20));  // 16 MB

  f32_to_f16_kernel<<<(BT_ * C_) / 256, 256, 0, stream>>>(x, X16, BT_ * C_);
  f32_to_f16_kernel<<<(F_ * C_) / 256, 256, 0, stream>>>(w_qkv, WQ16, F_ * C_);
  f32_to_f16_kernel<<<(C_ * C_) / 256, 256, 0, stream>>>(w_out, WO16, C_ * C_);

  // QKV = X @ Wqkv^T : M=8192, N=3072, K=1024 -> (64 x 48) blocks
  gemm_lds<true><<<64 * 48, 256, 0, stream>>>(X16, WQ16, QKV16, nullptr, C_, F_);

  v_transpose_kernel<<<(B_ * H_ * DH_ * T_) / 256, 256, 0, stream>>>(QKV16, VT16);

  // one wave per (b,h,qtile): 16*16*32 = 8192 waves
  attn_kernel<<<(B_ * H_ * (T_ / 16)) / 4, 128, 0, stream>>>(QKV16, VT16, kpm, O16);

  // out = O @ Wout^T + b : M=8192, N=1024, K=1024 -> (64 x 16) blocks
  gemm_lds<false><<<64 * 16, 256, 0, stream>>>(O16, WO16, out, b_out, C_, C_);
}
